// GPS_82274393522696
// MI455X (gfx1250) — compile-verified, matrix operand order checked by reading, not run on hardware
//
#include <hip/hip_runtime.h>
#include <cstdint>

// ---------------------------------------------------------------- constants
#define NB    128                 // graphs
#define NPG   128                 // nodes per graph
#define DIM   256                 // embedding dim
#define NL    5                   // layers
#define NH    4                   // heads
#define DH    64                  // head dim
#define NDRAW 128                 // raw node feature dim
#define EDRAW 64                  // raw edge feature dim
#define NNODE (NB * NPG)          // 16384
#define NEDGE (NNODE * 16)        // 262144

// ---------------------------------------------------------------- types
typedef __bf16  bf16;
typedef bf16    v16bf __attribute__((ext_vector_type(16)));
typedef float   v8f   __attribute__((ext_vector_type(8)));
typedef unsigned int v4u __attribute__((ext_vector_type(4)));

union Frag {            // one WMMA 16-bit operand: 16 bf16 per lane = 8 VGPRs
    v16bf v;
    v4u   q[2];
};

__device__ __forceinline__ v8f v8f_zero() {
    v8f z;
#pragma unroll
    for (int i = 0; i < 8; ++i) z[i] = 0.0f;
    return z;
}

// float -> bf16 (round to nearest even), kept as raw u16 so LDS tiles are POD
__device__ __forceinline__ unsigned short f2bf(float x) {
    unsigned int u = __float_as_uint(x);
    u += 0x7FFFu + ((u >> 16) & 1u);
    return (unsigned short)(u >> 16);
}

__device__ __forceinline__ float gelu_exact(float x) {
    return 0.5f * x * (1.0f + erff(x * 0.70710678118654752f));
}

// ============================================================== GEMM (WMMA)
// C[M,N] = act(A[M,K] @ W[N,K]^T + bias) (+ res)      all f32 in HBM,
// bf16 in LDS, f32 WMMA accumulate.
// 256 threads = 8 waves, 128x128 block tile, K stepped by 32.
// Wave w owns C rows [16w,16w+16) x 128 cols: 1 A frag + 8 B frags ->
// 8 v_wmma per K step, 64 f32 accumulator VGPRs per lane.
// ACT: 0=none 1=GELU 2=ReLU (compile-time, straight-line epilogue)
template <int ACT, bool HAS_RES, bool HAS_BIAS>
__global__ __launch_bounds__(256) void gemm_wmma_bf16(
    const float* __restrict__ A, const float* __restrict__ W,
    const float* __restrict__ bias, const float* __restrict__ res,
    float* __restrict__ C, int M, int K, int N)
{
    __shared__ __align__(16) unsigned short As[128][32];  // 8 KB, bf16 A tile
    __shared__ __align__(16) unsigned short Bs[128][32];  // 8 KB, bf16 W tile

    const int tid  = threadIdx.x;
    const int wave = tid >> 5;
    const int lane = tid & 31;
    const long m0  = (long)blockIdx.x * 128;
    const int  n0  = blockIdx.y * 128;

    v8f acc[8];
#pragma unroll
    for (int i = 0; i < 8; ++i) acc[i] = v8f_zero();

    const int fr = lane & 15;            // row within 16-row fragment
    const int kk = (lane >> 4) * 8;      // K sub-chunk (lanes 0-15 vs 16-31)

    for (int k0 = 0; k0 < K; k0 += 32) {
        // cooperative f32->bf16 tile load: 128 rows x 32 cols, A and W
        {
            const int r  = tid >> 1;           // 0..127
            const int h4 = (tid & 1) * 4;      // which 4 float4's of the row
            const float4* ap = reinterpret_cast<const float4*>(A + (m0 + r) * (long)K + k0);
            const float4* wp = reinterpret_cast<const float4*>(W + ((long)(n0 + r)) * K + k0);
            if (k0 + 32 < K) {   // prefetch next K tile -> global_prefetch_b8
                __builtin_prefetch(reinterpret_cast<const char*>(ap) + 128, 0, 1);
                __builtin_prefetch(reinterpret_cast<const char*>(wp) + 128, 0, 1);
            }
#pragma unroll
            for (int j = 0; j < 4; ++j) {
                float4 av = ap[h4 + j];
                float4 wv = wp[h4 + j];
                const int c = (h4 + j) * 4;
                As[r][c + 0] = f2bf(av.x); As[r][c + 1] = f2bf(av.y);
                As[r][c + 2] = f2bf(av.z); As[r][c + 3] = f2bf(av.w);
                Bs[r][c + 0] = f2bf(wv.x); Bs[r][c + 1] = f2bf(wv.y);
                Bs[r][c + 2] = f2bf(wv.z); Bs[r][c + 3] = f2bf(wv.w);
            }
        }
        __syncthreads();

        Frag a;
        a.q[0] = *reinterpret_cast<const v4u*>(&As[wave * 16 + fr][kk]);
        a.q[1] = *reinterpret_cast<const v4u*>(&As[wave * 16 + fr][kk + 16]);
#pragma unroll
        for (int nt = 0; nt < 8; ++nt) {
            Frag b;
            b.q[0] = *reinterpret_cast<const v4u*>(&Bs[nt * 16 + fr][kk]);
            b.q[1] = *reinterpret_cast<const v4u*>(&Bs[nt * 16 + fr][kk + 16]);
            acc[nt] = __builtin_amdgcn_wmma_f32_16x16x32_bf16(
                false, a.v, false, b.v, (short)0, acc[nt], false, false);
        }
        __syncthreads();
    }

    // epilogue: C/D layout -> lane covers col n0+nt*16+(lane&15),
    // rows m0+wave*16+8*(lane>>4)+v ; all modes compile-time straight-line
    const long rb = m0 + wave * 16 + ((lane >> 4) << 3);
#pragma unroll
    for (int nt = 0; nt < 8; ++nt) {
        const int n  = n0 + nt * 16 + fr;
        const float bv = HAS_BIAS ? bias[n] : 0.0f;
#pragma unroll
        for (int v = 0; v < 8; ++v) {
            float xv = acc[nt][v] + bv;
            if (ACT == 1)      xv = gelu_exact(xv);
            else if (ACT == 2) xv = fmaxf(xv, 0.0f);
            const long m = rb + v;
            if (HAS_RES) xv += res[m * N + n];
            C[m * N + n] = xv;
        }
    }
}

// ============================================== GINE message + scatter-add
// agg (pre-filled with h) += sum_e relu(h[src]+e)  via f32 global atomics
__global__ __launch_bounds__(256) void gine_scatter(
    const float* __restrict__ h, const float* __restrict__ e,
    const int* __restrict__ src, const int* __restrict__ dst,
    float* __restrict__ agg)
{
    const long idx = (long)blockIdx.x * 256 + threadIdx.x;
    const long tot = (long)NEDGE * (DIM / 4);
    if (idx >= tot) return;
    const int eid = (int)(idx >> 6);          // DIM/4 == 64
    const int c4  = ((int)idx & 63) * 4;
    const int s = src[eid];
    const int d = dst[eid];
    const float4 hv = *reinterpret_cast<const float4*>(h + (long)s * DIM + c4);
    const float4 ev = *reinterpret_cast<const float4*>(e + (long)eid * DIM + c4);
    float4 m;
    m.x = fmaxf(hv.x + ev.x, 0.0f);
    m.y = fmaxf(hv.y + ev.y, 0.0f);
    m.z = fmaxf(hv.z + ev.z, 0.0f);
    m.w = fmaxf(hv.w + ev.w, 0.0f);
    float* base = agg + (long)d * DIM + c4;
    atomicAdd(base + 0, m.x);
    atomicAdd(base + 1, m.y);
    atomicAdd(base + 2, m.z);
    atomicAdd(base + 3, m.w);
}

// ================================================= per-graph LayerNorm
// normalize over all NPG*DIM elements of one graph; gamma/beta per feature
__global__ __launch_bounds__(256) void graph_ln(
    const float* __restrict__ x, const float* __restrict__ gamma,
    const float* __restrict__ beta, float* __restrict__ y)
{
    __shared__ float s1[256];
    __shared__ float s2[256];
    const int b = blockIdx.x, t = threadIdx.x;
    const long base = (long)b * NPG * DIM;
    float lsum = 0.0f, lsq = 0.0f;
    for (int i = t; i < NPG * DIM; i += 256) {
        const float v = x[base + i];
        lsum += v; lsq += v * v;
    }
    s1[t] = lsum; s2[t] = lsq;
    __syncthreads();
    for (int off = 128; off > 0; off >>= 1) {
        if (t < off) { s1[t] += s1[t + off]; s2[t] += s2[t + off]; }
        __syncthreads();
    }
    const float inv_n = 1.0f / (float)(NPG * DIM);
    const float mu  = s1[0] * inv_n;
    const float var = s2[0] * inv_n - mu * mu;
    const float inv = rsqrtf(var + 1e-5f);
    for (int i = t; i < NPG * DIM; i += 256) {
        const int d = i & (DIM - 1);
        y[base + i] = (x[base + i] - mu) * inv * gamma[d] + beta[d];
    }
}

// ====================================== attention, one block per (graph,head)
// S = softmax(Q K^T / 8) ; O = S V, all matmuls through bf16 WMMA.
// LDS: QK 32KB (reused as bf16 P), Vt 16KB, S 64KB  => 112KB (< 320KB WGP LDS)
__global__ __launch_bounds__(128) void attn_wmma(
    const float* __restrict__ qkv,    // [NNODE, 3*DIM]
    float* __restrict__ out)          // [NNODE, DIM]
{
    __shared__ __align__(16) unsigned short QK[2][NPG][DH]; // Q then K (bf16)
    __shared__ __align__(16) unsigned short Vt[DH][NPG];    // V transposed (bf16)
    __shared__ __align__(16) float S[NPG][NPG];             // scores (f32)

    const int bh   = blockIdx.x;
    const int g    = bh >> 2;     // graph   (NH == 4)
    const int hd   = bh & 3;      // head
    const int tid  = threadIdx.x; // 128 threads = 4 waves
    const int wave = tid >> 5;
    const int lane = tid & 31;
    const int fr   = lane & 15;
    const int kk   = (lane >> 4) * 8;

    // ---- load q/k/v of node `tid`, convert to bf16
    {
        const long node = (long)g * NPG + tid;
        const float* qp = qkv + node * (3 * DIM) + hd * DH;
#pragma unroll
        for (int d4 = 0; d4 < DH; d4 += 4) {
            float4 q4 = *reinterpret_cast<const float4*>(qp + d4);
            float4 k4 = *reinterpret_cast<const float4*>(qp + DIM + d4);
            float4 v4 = *reinterpret_cast<const float4*>(qp + 2 * DIM + d4);
            QK[0][tid][d4 + 0] = f2bf(q4.x); QK[0][tid][d4 + 1] = f2bf(q4.y);
            QK[0][tid][d4 + 2] = f2bf(q4.z); QK[0][tid][d4 + 3] = f2bf(q4.w);
            QK[1][tid][d4 + 0] = f2bf(k4.x); QK[1][tid][d4 + 1] = f2bf(k4.y);
            QK[1][tid][d4 + 2] = f2bf(k4.z); QK[1][tid][d4 + 3] = f2bf(k4.w);
            Vt[d4 + 0][tid] = f2bf(v4.x);    Vt[d4 + 1][tid] = f2bf(v4.y);
            Vt[d4 + 2][tid] = f2bf(v4.z);    Vt[d4 + 3][tid] = f2bf(v4.w);
        }
    }
    __syncthreads();

    // ---- phase 1: S = (Q @ K^T) * 1/sqrt(DH); wave w owns query rows [32w,32w+32)
#pragma unroll
    for (int mt = 0; mt < 2; ++mt) {
        const int r0 = wave * 32 + mt * 16;
        Frag a0, a1;
        a0.q[0] = *reinterpret_cast<const v4u*>(&QK[0][r0 + fr][kk]);
        a0.q[1] = *reinterpret_cast<const v4u*>(&QK[0][r0 + fr][kk + 16]);
        a1.q[0] = *reinterpret_cast<const v4u*>(&QK[0][r0 + fr][32 + kk]);
        a1.q[1] = *reinterpret_cast<const v4u*>(&QK[0][r0 + fr][32 + kk + 16]);
        for (int nt = 0; nt < 8; ++nt) {
            Frag b0, b1;
            b0.q[0] = *reinterpret_cast<const v4u*>(&QK[1][nt * 16 + fr][kk]);
            b0.q[1] = *reinterpret_cast<const v4u*>(&QK[1][nt * 16 + fr][kk + 16]);
            b1.q[0] = *reinterpret_cast<const v4u*>(&QK[1][nt * 16 + fr][32 + kk]);
            b1.q[1] = *reinterpret_cast<const v4u*>(&QK[1][nt * 16 + fr][32 + kk + 16]);
            v8f acc = v8f_zero();
            acc = __builtin_amdgcn_wmma_f32_16x16x32_bf16(
                false, a0.v, false, b0.v, (short)0, acc, false, false);
            acc = __builtin_amdgcn_wmma_f32_16x16x32_bf16(
                false, a1.v, false, b1.v, (short)0, acc, false, false);
            const int rb = r0 + ((lane >> 4) << 3);
#pragma unroll
            for (int v = 0; v < 8; ++v) S[rb + v][nt * 16 + fr] = acc[v] * 0.125f;
        }
    }
    __syncthreads();

    // ---- phase 2: row softmax, pack bf16 P over the dead Q/K region
    unsigned short (*P)[NPG] = reinterpret_cast<unsigned short (*)[NPG]>(&QK[0][0][0]);
    {
        const int r = tid;
        float mx = -3.4e38f;
        for (int j = 0; j < NPG; ++j) mx = fmaxf(mx, S[r][j]);
        float sum = 0.0f;
        for (int j = 0; j < NPG; ++j) sum += __expf(S[r][j] - mx);
        const float inv = 1.0f / sum;
        for (int j = 0; j < NPG; ++j) P[r][j] = f2bf(__expf(S[r][j] - mx) * inv);
    }
    __syncthreads();

    // ---- phase 3: O = P @ V   (M=128, N=64, K=128)
#pragma unroll
    for (int mt = 0; mt < 2; ++mt) {
        const int r0 = wave * 32 + mt * 16;
#pragma unroll
        for (int nt = 0; nt < 4; ++nt) {
            v8f acc = v8f_zero();
#pragma unroll
            for (int k0 = 0; k0 < NPG; k0 += 32) {
                Frag a, b;
                a.q[0] = *reinterpret_cast<const v4u*>(&P[r0 + fr][k0 + kk]);
                a.q[1] = *reinterpret_cast<const v4u*>(&P[r0 + fr][k0 + kk + 16]);
                b.q[0] = *reinterpret_cast<const v4u*>(&Vt[nt * 16 + fr][k0 + kk]);
                b.q[1] = *reinterpret_cast<const v4u*>(&Vt[nt * 16 + fr][k0 + kk + 16]);
                acc = __builtin_amdgcn_wmma_f32_16x16x32_bf16(
                    false, a.v, false, b.v, (short)0, acc, false, false);
            }
            const int rb = r0 + ((lane >> 4) << 3);
            const int n  = nt * 16 + fr;
#pragma unroll
            for (int v = 0; v < 8; ++v) {
                const long node = (long)g * NPG + rb + v;
                out[node * DIM + hd * DH + n] = acc[v];
            }
        }
    }
}

// ---------------------------------------------------------------- small ops
__global__ __launch_bounds__(256) void vec_add4(
    const float* __restrict__ a, const float* __restrict__ b,
    float* __restrict__ c, long n4)
{
    const long i = (long)blockIdx.x * 256 + threadIdx.x;
    if (i >= n4) return;
    const float4 av = reinterpret_cast<const float4*>(a)[i];
    const float4 bv = reinterpret_cast<const float4*>(b)[i];
    float4 cv;
    cv.x = av.x + bv.x; cv.y = av.y + bv.y;
    cv.z = av.z + bv.z; cv.w = av.w + bv.w;
    reinterpret_cast<float4*>(c)[i] = cv;
}

__global__ __launch_bounds__(256) void pool_sum(
    const float* __restrict__ h, float* __restrict__ glob)
{
    const int b = blockIdx.x, d = threadIdx.x;   // DIM == 256 threads
    float s = 0.0f;
    for (int i = 0; i < NPG; ++i) s += h[((long)b * NPG + i) * DIM + d];
    glob[b * DIM + d] = s;
}

// ================================================================== driver
extern "C" void kernel_launch(void* const* d_in, const int* in_sizes, int n_in,
                              void* d_out, int out_size, void* d_ws, size_t ws_size,
                              hipStream_t stream)
{
    (void)in_sizes; (void)n_in; (void)out_size; (void)ws_size;

    const float* x          = (const float*)d_in[0];
    const float* edge_attr  = (const float*)d_in[1];
    const int*   edge_index = (const int*)d_in[2];
    const float* node_emb_w = (const float*)d_in[4];
    const float* edge_emb_w = (const float*)d_in[5];
    const float* conv_w1    = (const float*)d_in[6];
    const float* conv_b1    = (const float*)d_in[7];
    const float* conv_w2    = (const float*)d_in[8];
    const float* conv_b2    = (const float*)d_in[9];
    const float* attn_in_w  = (const float*)d_in[10];
    const float* attn_in_b  = (const float*)d_in[11];
    const float* attn_out_w = (const float*)d_in[12];
    const float* attn_out_b = (const float*)d_in[13];
    const float* n1_g = (const float*)d_in[14];
    const float* n1_b = (const float*)d_in[15];
    const float* n2_g = (const float*)d_in[16];
    const float* n2_b = (const float*)d_in[17];
    const float* n3_g = (const float*)d_in[18];
    const float* n3_b = (const float*)d_in[19];
    const float* mlp_w1 = (const float*)d_in[20];
    const float* mlp_b1 = (const float*)d_in[21];
    const float* mlp_w2 = (const float*)d_in[22];
    const float* mlp_b2 = (const float*)d_in[23];

    const int* srcI = edge_index;           // row 0
    const int* dstI = edge_index + NEDGE;   // row 1

    // workspace carve-up (all f32)
    float* p = reinterpret_cast<float*>(d_ws);
    auto carve = [&](size_t n) { float* r = p; p += n; return r; };
    float* h      = carve((size_t)NNODE * DIM);
    float* ebuf   = carve((size_t)NEDGE * DIM);
    float* z      = carve((size_t)NNODE * DIM);
    float* z1     = carve((size_t)NNODE * DIM);
    float* z2     = carve((size_t)NNODE * DIM);
    float* h1     = carve((size_t)NNODE * DIM);
    float* qkvb   = carve((size_t)NNODE * 3 * DIM);
    float* attnA  = carve((size_t)NNODE * DIM);
    float* h2in   = carve((size_t)NNODE * DIM);
    float* h2     = carve((size_t)NNODE * DIM);
    float* hsum   = carve((size_t)NNODE * DIM);
    float* mlpH   = carve((size_t)NNODE * 2 * DIM);
    float* outpre = carve((size_t)NNODE * DIM);

    const dim3 blk256(256);
    auto grid_of = [](int M, int N) { return dim3((unsigned)(M / 128), (unsigned)(N / 128)); };

    // embeddings (no bias, no activation, no residual)
    gemm_wmma_bf16<0, false, false><<<grid_of(NNODE, DIM), blk256, 0, stream>>>(
        x, node_emb_w, nullptr, nullptr, h, NNODE, NDRAW, DIM);
    gemm_wmma_bf16<0, false, false><<<grid_of(NEDGE, DIM), blk256, 0, stream>>>(
        edge_attr, edge_emb_w, nullptr, nullptr, ebuf, NEDGE, EDRAW, DIM);

    const long scat_thr = (long)NEDGE * (DIM / 4);
    for (int l = 0; l < NL; ++l) {
        const float* w1  = conv_w1    + (size_t)l * DIM * DIM;
        const float* b1  = conv_b1    + (size_t)l * DIM;
        const float* w2  = conv_w2    + (size_t)l * DIM * DIM;
        const float* b2  = conv_b2    + (size_t)l * DIM;
        const float* inw = attn_in_w  + (size_t)l * 3 * DIM * DIM;
        const float* inb = attn_in_b  + (size_t)l * 3 * DIM;
        const float* ow  = attn_out_w + (size_t)l * DIM * DIM;
        const float* ob  = attn_out_b + (size_t)l * DIM;
        const float* mw1 = mlp_w1     + (size_t)l * 2 * DIM * DIM;
        const float* mb1 = mlp_b1     + (size_t)l * 2 * DIM;
        const float* mw2 = mlp_w2     + (size_t)l * DIM * 2 * DIM;
        const float* mb2 = mlp_b2     + (size_t)l * DIM;

        // local branch: z = h + sum relu(h[src]+e); Linear-GELU-Linear; LN1
        hipMemcpyAsync(z, h, (size_t)NNODE * DIM * sizeof(float),
                       hipMemcpyDeviceToDevice, stream);
        gine_scatter<<<(unsigned)((scat_thr + 255) / 256), 256, 0, stream>>>(
            h, ebuf, srcI, dstI, z);
        gemm_wmma_bf16<1, false, true><<<grid_of(NNODE, DIM), blk256, 0, stream>>>(
            z, w1, b1, nullptr, z1, NNODE, DIM, DIM);                     // GELU
        gemm_wmma_bf16<0, true, true><<<grid_of(NNODE, DIM), blk256, 0, stream>>>(
            z1, w2, b2, h, z2, NNODE, DIM, DIM);                          // + h
        graph_ln<<<NB, 256, 0, stream>>>(z2, n1_g + (size_t)l * DIM,
                                         n1_b + (size_t)l * DIM, h1);

        // global branch: attention on h; LN2
        gemm_wmma_bf16<0, false, true><<<grid_of(NNODE, 3 * DIM), blk256, 0, stream>>>(
            h, inw, inb, nullptr, qkvb, NNODE, DIM, 3 * DIM);
        attn_wmma<<<NB * NH, 128, 0, stream>>>(qkvb, attnA);
        gemm_wmma_bf16<0, true, true><<<grid_of(NNODE, DIM), blk256, 0, stream>>>(
            attnA, ow, ob, h, h2in, NNODE, DIM, DIM);                     // + h
        graph_ln<<<NB, 256, 0, stream>>>(h2in, n2_g + (size_t)l * DIM,
                                         n2_b + (size_t)l * DIM, h2);

        // combine + MLP + LN3 -> h (next layer input)
        vec_add4<<<(unsigned)(((long)NNODE * DIM / 4 + 255) / 256), 256, 0, stream>>>(
            h1, h2, hsum, (long)NNODE * DIM / 4);
        gemm_wmma_bf16<2, false, true><<<grid_of(NNODE, 2 * DIM), blk256, 0, stream>>>(
            hsum, mw1, mb1, nullptr, mlpH, NNODE, DIM, 2 * DIM);          // ReLU
        gemm_wmma_bf16<0, true, true><<<grid_of(NNODE, DIM), blk256, 0, stream>>>(
            mlpH, mw2, mb2, hsum, outpre, NNODE, 2 * DIM, DIM);           // + hsum
        graph_ln<<<NB, 256, 0, stream>>>(outpre, n3_g + (size_t)l * DIM,
                                         n3_b + (size_t)l * DIM, h);
    }

    // outputs: h [N,D] then global_add_pool [B,D]
    float* outF = reinterpret_cast<float*>(d_out);
    hipMemcpyAsync(outF, h, (size_t)NNODE * DIM * sizeof(float),
                   hipMemcpyDeviceToDevice, stream);
    pool_sum<<<NB, DIM, 0, stream>>>(h, outF + (size_t)NNODE * DIM);
}
